// MyMultiHeadAttention_73160472920691
// MI455X (gfx1250) — compile-verified, matrix operand order checked by reading, not run on hardware
//
#include <hip/hip_runtime.h>

// ---------------------------------------------------------------------------
// CDNA5 (gfx1250) fused MHA: bf16 WMMA GEMMs + flash-attention, with the
// Tensor Data Mover (TDM) doing all global->LDS tile staging asynchronously
// (double-buffered, TENSORcnt-tracked) so the VALU/VMEM path is free for
// WMMA operand loads and softmax.
//
// B=4, S=2048, E=1024, H=16, DH=64. split_heads is a pure reshape, so each
// head is a contiguous (2048 x 64) row-major chunk of the projected buffer.
//
// Fragment codegen: in the CDNA5 16-bit A/B wave32 layouts, each lane's
// fragment is two contiguous 16-byte runs of one tile row:
//   k = k0 + half*8 + [0..7]   and   k0 + 16 + half*8 + [0..7]
// so tiles stored contraction-contiguous per lane-row load as two b128s.
// LDS leading dims padded (40 / 72 elements) via the TDM pad feature for
// conflict-free lane->bank spread.
// ---------------------------------------------------------------------------

typedef __attribute__((ext_vector_type(16))) __bf16 v16bf;
typedef __attribute__((ext_vector_type(8)))  float  v8f;
typedef __attribute__((ext_vector_type(4)))  unsigned int v4u;
typedef __attribute__((ext_vector_type(8)))  int v8i;
typedef __attribute__((ext_vector_type(4)))  int v4i;

#define WMMA_BF16(a, b, c) \
  __builtin_amdgcn_wmma_f32_16x16x32_bf16(false, (a), false, (b), (short)0, (c), false, false)

// Two contiguous b128 loads -> one 16x16x32 bf16 operand fragment.
static __device__ __forceinline__ v16bf frag_ld(const __bf16* rowbase, int half, int k0) {
  union { uint4 u[2]; v16bf v; } r;
  const uint4* p = (const uint4*)(rowbase + k0 + half * 8);
  r.u[0] = p[0];
  r.u[1] = p[2];  // +16 elements = +2 uint4
  return r.v;
}

// ---------------------------------------------------------------------------
// TDM 2-D tile load: bf16 tile (tile_w x tile_h) from global (row stride in
// elements) into LDS with per-row padding. Descriptor per ISA cdna5 §8.
// pad_interval code: 0=2,1=4,2=8,3=16,4=32... DWORDs between pads.
// pad_amount   code: n -> (n+1) DWORDs of padding.
// Issue from ONE wave only; completion via s_wait_tensorcnt.
// ---------------------------------------------------------------------------
static __device__ __forceinline__ void tdm_load_tile_bf16(
    const void* lds_dst, const __bf16* gsrc, int tile_w, int tile_h,
    long long row_stride_elems, int pad_interval_code, int pad_amount_code) {
  const unsigned long long ga = (unsigned long long)(uintptr_t)gsrc;
  v4u g0 = {0u, 0u, 0u, 0u};
  g0[0] = 1u;                                             // count = 1 (valid)
  g0[1] = (unsigned)(uintptr_t)lds_dst;                   // lds_addr (bytes)
  g0[2] = (unsigned)(ga & 0xffffffffull);                 // global_addr lo
  g0[3] = (unsigned)((ga >> 32) & 0x1ffffffull)           // global_addr hi
        | (2u << 30);                                     // type = 2 ("image")
  v8i g1 = {0, 0, 0, 0, 0, 0, 0, 0};
  g1[0] = (1 << 16)                                       // data_size = 2 bytes
        | (1 << 20)                                       // pad_enable
        | (pad_interval_code << 22) | (pad_amount_code << 25);
  g1[1] = (tile_w & 0xffff) << 16;                        // tensor_dim0 lo16
  g1[2] = ((tile_w >> 16) & 0xffff) | ((tile_h & 0xffff) << 16);  // d0 hi|d1 lo
  g1[3] = ((tile_h >> 16) & 0xffff) | ((tile_w & 0xffff) << 16);  // d1 hi|tile_dim0
  g1[4] = (tile_h & 0xffff);                              // tile_dim1 (tile_dim2=0)
  g1[5] = (int)(row_stride_elems & 0xffffffffll);         // dim0_stride lo32
  g1[6] = (int)((row_stride_elems >> 32) & 0xffffll);     // dim0_stride hi16
  const v4i z4 = {0, 0, 0, 0};
  const v8i z8 = {0, 0, 0, 0, 0, 0, 0, 0};
  __builtin_amdgcn_tensor_load_to_lds(g0, g1, z4, z4, z8, 0);
}

// ---------------------------------------------------------------------------
// f32 -> bf16 cast (x4 vectorized)
// ---------------------------------------------------------------------------
__global__ __launch_bounds__(256) void cast_f32_bf16(const float* __restrict__ src,
                                                     __bf16* __restrict__ dst, int n) {
  const int i = (blockIdx.x * 256 + threadIdx.x) * 4;
  if (i + 3 < n) {
    const float4 v = *(const float4*)(src + i);
    dst[i]     = (__bf16)v.x;
    dst[i + 1] = (__bf16)v.y;
    dst[i + 2] = (__bf16)v.z;
    dst[i + 3] = (__bf16)v.w;
  }
}

// ---------------------------------------------------------------------------
// Weight cast+transpose: W (K x N, f32) -> Wt (N x K, bf16). 32x32 LDS tiles.
// ---------------------------------------------------------------------------
__global__ __launch_bounds__(256) void cast_transpose_bf16(const float* __restrict__ src,
                                                           __bf16* __restrict__ dst,
                                                           int K, int N) {
  __shared__ __bf16 tile[32][33];
  const int k0 = blockIdx.x * 32, n0 = blockIdx.y * 32;
  const int t = threadIdx.x;
  {
    const int r = t >> 3, c = (t & 7) * 4;
    const float4 v = *(const float4*)(src + (size_t)(k0 + r) * N + n0 + c);
    tile[r][c + 0] = (__bf16)v.x;
    tile[r][c + 1] = (__bf16)v.y;
    tile[r][c + 2] = (__bf16)v.z;
    tile[r][c + 3] = (__bf16)v.w;
  }
  __syncthreads();
  const int n = t >> 3, kc = (t & 7) * 4;
  __bf16* o = dst + (size_t)(n0 + n) * K + k0 + kc;
  o[0] = tile[kc + 0][n];
  o[1] = tile[kc + 1][n];
  o[2] = tile[kc + 2][n];
  o[3] = tile[kc + 3][n];
}

// ---------------------------------------------------------------------------
// Per-head transpose: Vp head chunk (2048 keys x 64 dims) -> Vtg (64 x 2048).
// ---------------------------------------------------------------------------
__global__ __launch_bounds__(128) void transpose_heads(const __bf16* __restrict__ src,
                                                       __bf16* __restrict__ dst) {
  constexpr int S = 2048, DH = 64, LDT = 72;
  __shared__ alignas(16) __bf16 tile[64 * LDT];
  const int head = blockIdx.x, s0 = blockIdx.y * 64;
  const int t = threadIdx.x;
  const __bf16* sp = src + ((size_t)head * S + s0) * DH;
  __bf16* dp = dst + (size_t)head * DH * S;
#pragma unroll
  for (int i = 0; i < 4; ++i) {
    const int idx = t + 128 * i;
    const int r = idx >> 3, c = idx & 7;
    *(uint4*)(tile + r * LDT + c * 8) = *(const uint4*)(sp + r * DH + c * 8);
  }
  __syncthreads();
#pragma unroll
  for (int i = 0; i < 32; ++i) {
    const int idx = t + 128 * i;
    const int d = idx >> 6, s = idx & 63;
    dp[(size_t)d * S + s0 + s] = tile[s * LDT + d];
  }
}

// ---------------------------------------------------------------------------
// GEMM: out[M,N] = A[M,K](bf16) @ W[K,N] + bias[N], W given transposed (NxK).
// 256 thr = 8 waves; workgroup tile 256(M) x 64(N); wave = 32x64 (8 WMMA/step).
// TDM stages A (256x32 -> ld 40) and B (64x32 N-major -> ld 40), 2x buffered.
// ---------------------------------------------------------------------------
template <typename OutT>
__global__ __launch_bounds__(256) void gemm_bf16(const __bf16* __restrict__ A,
                                                 const __bf16* __restrict__ Wt,
                                                 const float* __restrict__ bias,
                                                 OutT* __restrict__ out,
                                                 int M, int N, int K) {
  constexpr int LDA = 40, LDB = 40;
  __shared__ alignas(16) __bf16 As[2][256 * LDA];
  __shared__ alignas(16) __bf16 Bs[2][64 * LDB];
  const int t = threadIdx.x, wave = t >> 5, lane = t & 31;
  const int half = lane >> 4, ml = lane & 15;
  const int row0 = blockIdx.x * 256, col0 = blockIdx.y * 64;

  auto issue = [&](int buf, int k0) {
    tdm_load_tile_bf16(As[buf], A + (size_t)row0 * K + k0, 32, 256, K, 3, 3);
    tdm_load_tile_bf16(Bs[buf], Wt + (size_t)col0 * K + k0, 32, 64, K, 3, 3);
  };

  v8f zero = {};
  v8f acc[2][4] = {{zero, zero, zero, zero}, {zero, zero, zero, zero}};

  if (wave == 0) { issue(0, 0); __builtin_amdgcn_s_wait_tensorcnt(0); }
  __syncthreads();

  const int nk = K / 32;
  for (int ik = 0; ik < nk; ++ik) {
    const int buf = ik & 1;
    if (wave == 0 && ik + 1 < nk) issue(buf ^ 1, (ik + 1) * 32);  // prefetch

    v16bf af[2];
#pragma unroll
    for (int mt = 0; mt < 2; ++mt)
      af[mt] = frag_ld(As[buf] + (wave * 32 + mt * 16 + ml) * LDA, half, 0);
#pragma unroll
    for (int tt = 0; tt < 4; ++tt) {
      const v16bf bf = frag_ld(Bs[buf] + (tt * 16 + ml) * LDB, half, 0);
      acc[0][tt] = WMMA_BF16(af[0], bf, acc[0][tt]);
      acc[1][tt] = WMMA_BF16(af[1], bf, acc[1][tt]);
    }

    if (wave == 0) __builtin_amdgcn_s_wait_tensorcnt(0);
    __syncthreads();
  }

#pragma unroll
  for (int tt = 0; tt < 4; ++tt) {
    const int col = col0 + tt * 16 + ml;
    const float bv = bias[col];
#pragma unroll
    for (int mt = 0; mt < 2; ++mt)
#pragma unroll
      for (int j = 0; j < 8; ++j) {
        const int row = row0 + wave * 32 + mt * 16 + half * 8 + j;
        out[(size_t)row * N + col] = (OutT)(acc[mt][tt][j] + bv);
      }
  }
}

// ---------------------------------------------------------------------------
// Flash attention over one head (contiguous 2048x64 bf16 chunk), V transposed.
// grid = (64 heads, 32 q-blocks); 128 threads = 4 waves; wave owns 16 q rows.
// TDM stages K (64x64 row-major) and V (64x64 from DHxS), 2x buffered, ld 72.
// ---------------------------------------------------------------------------
__global__ __launch_bounds__(128) void attn_kernel(const __bf16* __restrict__ Qp,
                                                   const __bf16* __restrict__ Kp,
                                                   const __bf16* __restrict__ Vtg,
                                                   __bf16* __restrict__ Zp) {
  constexpr int S = 2048, DH = 64, LDT = 72;
  constexpr float NEGV = -1.0e11f;
  __shared__ alignas(16) __bf16 Kt[2][64 * LDT];  // [key][dh]
  __shared__ alignas(16) __bf16 Vt[2][64 * LDT];  // [dh][key]
  __shared__ alignas(16) __bf16 Pt[4][16 * LDT];  // per-wave P staging

  const int head = blockIdx.x;  // b*H + h
  const int qblk = blockIdx.y;
  const int t = threadIdx.x, wave = t >> 5, lane = t & 31;
  const int half = lane >> 4, ml = lane & 15;
  const size_t hb = (size_t)head * S * DH;
  const __bf16* Qh = Qp + hb;
  const __bf16* Kh = Kp + hb;
  const __bf16* Vh = Vtg + hb;  // DH x S
  __bf16* Zh = Zp + hb;
  const int q0 = qblk * 64 + wave * 16;

  auto issueKV = [&](int buf, int kb0) {
    tdm_load_tile_bf16(Kt[buf], Kh + (size_t)kb0 * DH, 64, 64, DH, 4, 3);
    tdm_load_tile_bf16(Vt[buf], Vh + kb0, 64, 64, S, 4, 3);
  };

  if (wave == 0) issueKV(0, 0);
  const v16bf aq0 = frag_ld(Qh + (size_t)(q0 + ml) * DH, half, 0);
  const v16bf aq1 = frag_ld(Qh + (size_t)(q0 + ml) * DH, half, 32);
  if (wave == 0) __builtin_amdgcn_s_wait_tensorcnt(0);
  __syncthreads();

  v8f zero = {};
  v8f o[4] = {zero, zero, zero, zero};
  float mrun[8], lrun[8];
#pragma unroll
  for (int j = 0; j < 8; ++j) { mrun[j] = -3.0e38f; lrun[j] = 0.0f; }

  const float scale = 0.125f;  // 1/sqrt(DH)
  const int nblocks = qblk + 1;

  for (int blk = 0; blk < nblocks; ++blk) {
    const int buf = blk & 1;
    const int kb0 = blk * 64;
    if (wave == 0 && blk + 1 < nblocks) issueKV(buf ^ 1, kb0 + 64);  // prefetch

    // S = (Q @ K^T) * scale : B[k=dh][n=key] = Kt[key][dh] (contiguous runs)
    v8f s[4] = {zero, zero, zero, zero};
#pragma unroll
    for (int tt = 0; tt < 4; ++tt) {
      const __bf16* krow = Kt[buf] + (tt * 16 + ml) * LDT;
      s[tt] = WMMA_BF16(aq0, frag_ld(krow, half, 0), s[tt]);
      s[tt] = WMMA_BF16(aq1, frag_ld(krow, half, 32), s[tt]);
    }
#pragma unroll
    for (int tt = 0; tt < 4; ++tt)
#pragma unroll
      for (int j = 0; j < 8; ++j) s[tt][j] *= scale;

    if (blk == qblk) {  // causal mask on diagonal block
#pragma unroll
      for (int tt = 0; tt < 4; ++tt) {
        const int col = kb0 + tt * 16 + ml;
#pragma unroll
        for (int j = 0; j < 8; ++j) {
          const int row = q0 + half * 8 + j;
          if (col > row) s[tt][j] = NEGV;
        }
      }
    }

    // Online softmax (row reductions across the 16-lane column groups)
    float alpha[8], rs[8];
#pragma unroll
    for (int j = 0; j < 8; ++j) {
      float mx = fmaxf(fmaxf(s[0][j], s[1][j]), fmaxf(s[2][j], s[3][j]));
      mx = fmaxf(mx, __shfl_xor(mx, 1, 32));
      mx = fmaxf(mx, __shfl_xor(mx, 2, 32));
      mx = fmaxf(mx, __shfl_xor(mx, 4, 32));
      mx = fmaxf(mx, __shfl_xor(mx, 8, 32));
      const float mnew = fmaxf(mrun[j], mx);
      alpha[j] = __expf(mrun[j] - mnew);
      mrun[j] = mnew;
      rs[j] = 0.0f;
    }
#pragma unroll
    for (int tt = 0; tt < 4; ++tt)
#pragma unroll
      for (int j = 0; j < 8; ++j) {
        const float p = __expf(s[tt][j] - mrun[j]);
        s[tt][j] = p;
        rs[j] += p;
      }
#pragma unroll
    for (int j = 0; j < 8; ++j) {
      float r = rs[j];
      r += __shfl_xor(r, 1, 32);
      r += __shfl_xor(r, 2, 32);
      r += __shfl_xor(r, 4, 32);
      r += __shfl_xor(r, 8, 32);
      lrun[j] = lrun[j] * alpha[j] + r;
    }

    // C-layout -> A-layout via per-wave LDS tile (same-wave DS ops in-order)
    __bf16* pw = &Pt[wave][0];
#pragma unroll
    for (int tt = 0; tt < 4; ++tt)
#pragma unroll
      for (int j = 0; j < 8; ++j)
        pw[(half * 8 + j) * LDT + tt * 16 + ml] = (__bf16)s[tt][j];

#pragma unroll
    for (int tt = 0; tt < 4; ++tt)
#pragma unroll
      for (int j = 0; j < 8; ++j) o[tt][j] *= alpha[j];

    // O += P @ V : B[k=key][n=dh] = Vt[dh][key] (contiguous runs)
    const v16bf ap0 = frag_ld(pw + ml * LDT, half, 0);
    const v16bf ap1 = frag_ld(pw + ml * LDT, half, 32);
#pragma unroll
    for (int tt = 0; tt < 4; ++tt) {
      const __bf16* vrow = Vt[buf] + (tt * 16 + ml) * LDT;
      o[tt] = WMMA_BF16(ap0, frag_ld(vrow, half, 0), o[tt]);
      o[tt] = WMMA_BF16(ap1, frag_ld(vrow, half, 32), o[tt]);
    }

    if (wave == 0) __builtin_amdgcn_s_wait_tensorcnt(0);
    __syncthreads();
  }

  // Z = O / l  (row-major head chunk == inverse reshape)
#pragma unroll
  for (int tt = 0; tt < 4; ++tt)
#pragma unroll
    for (int j = 0; j < 8; ++j) {
      const float inv = 1.0f / lrun[j];
      Zh[(size_t)(q0 + half * 8 + j) * DH + tt * 16 + ml] = (__bf16)(o[tt][j] * inv);
    }
}

// ---------------------------------------------------------------------------
extern "C" void kernel_launch(void* const* d_in, const int* in_sizes, int n_in,
                              void* d_out, int out_size, void* d_ws, size_t ws_size,
                              hipStream_t stream) {
  (void)in_sizes; (void)n_in; (void)out_size; (void)ws_size;
  const float* x  = (const float*)d_in[0];
  const float* Wq = (const float*)d_in[1];
  const float* bq = (const float*)d_in[2];
  const float* Wk = (const float*)d_in[3];
  const float* bk = (const float*)d_in[4];
  const float* Wv = (const float*)d_in[5];
  const float* bv = (const float*)d_in[6];
  const float* Wo = (const float*)d_in[7];
  const float* bo = (const float*)d_in[8];
  float* out = (float*)d_out;

  const int Bb = 4, S = 2048, E = 1024;
  const int M = Bb * S;  // 8192

  // Workspace layout (bf16), 104 MB total.
  const size_t MB = (size_t)1 << 20;
  char* w = (char*)d_ws;
  __bf16* xh  = (__bf16*)(w);             // 16 MB
  __bf16* Wqt = (__bf16*)(w + 16 * MB);   // 2 MB each (transposed)
  __bf16* Wkt = (__bf16*)(w + 18 * MB);
  __bf16* Wvt = (__bf16*)(w + 20 * MB);
  __bf16* Wot = (__bf16*)(w + 22 * MB);
  __bf16* Qp  = (__bf16*)(w + 24 * MB);   // 16 MB each
  __bf16* Kp  = (__bf16*)(w + 40 * MB);
  __bf16* Vp  = (__bf16*)(w + 56 * MB);
  __bf16* Zp  = (__bf16*)(w + 72 * MB);
  __bf16* Vtg = (__bf16*)(w + 88 * MB);   // V transposed per head

  cast_f32_bf16<<<(M * E) / 1024, 256, 0, stream>>>(x, xh, M * E);
  const dim3 tg(E / 32, E / 32);
  cast_transpose_bf16<<<tg, 256, 0, stream>>>(Wq, Wqt, E, E);
  cast_transpose_bf16<<<tg, 256, 0, stream>>>(Wk, Wkt, E, E);
  cast_transpose_bf16<<<tg, 256, 0, stream>>>(Wv, Wvt, E, E);
  cast_transpose_bf16<<<tg, 256, 0, stream>>>(Wo, Wot, E, E);

  const dim3 gg(M / 256, E / 64);
  gemm_bf16<__bf16><<<gg, 256, 0, stream>>>(xh, Wqt, bq, Qp, M, E, E);
  gemm_bf16<__bf16><<<gg, 256, 0, stream>>>(xh, Wkt, bk, Kp, M, E, E);
  gemm_bf16<__bf16><<<gg, 256, 0, stream>>>(xh, Wvt, bv, Vp, M, E, E);

  transpose_heads<<<dim3(64, 32), 128, 0, stream>>>(Vp, Vtg);

  attn_kernel<<<dim3(64, 32), 128, 0, stream>>>(Qp, Kp, Vtg, Zp);

  gemm_bf16<float><<<gg, 256, 0, stream>>>(Zp, Wot, bo, out, M, E, E);
}
